// MultiHeadAttentionWithRoPE_32933809226177
// MI455X (gfx1250) — matrix operand order repaired and checked by reading, not compiled, not run on hardware
//
#include <hip/hip_runtime.h>

// MultiHeadAttention + RoPE for MI455X (gfx1250), bf16 WMMA pipeline with
// async global->LDS copies (ASYNCcnt) and double-buffered GEMM mainloops.
// B=2, S=2048, D=1024, H=16, Dh=64.
// Workspace layout (bytes):
//   [0,8M)    x in bf16                  [8M,10M)  Wq bf16
//   [10M,12M) Wk bf16                    [12M,14M) Wv bf16
//   [14M,16M) Wo bf16
//   [16M,24M) Q  [B,H,S,Dh] bf16 (RoPE'd)
//   [24M,32M) K  [B,H,S,Dh] bf16 (RoPE'd)
//   [32M,40M) V  [B,H,S,Dh] bf16
//   [40M,48M) ctx [B,S,D] bf16
// Requires ws_size >= 48 MiB.

#define B_  2
#define S_  2048
#define D_  1024
#define H_  16
#define DH_ 64

typedef unsigned short u16;
typedef __attribute__((ext_vector_type(16))) __bf16 v16bf;
typedef __attribute__((ext_vector_type(8)))  float  v8f;

static __device__ __forceinline__ u16 bf16_of(float f) {
  unsigned u = __float_as_uint(f);
  unsigned r = u + 0x7FFFu + ((u >> 16) & 1u);   // round-to-nearest-even
  return (u16)(r >> 16);
}

static __device__ __forceinline__ v16bf frag_of(uint4 a, uint4 b) {
  union { uint4 q[2]; v16bf v; } t;
  t.q[0] = a; t.q[1] = b;
  return t.v;
}

static __device__ __forceinline__ v8f vzero8() {
  v8f z = {0.f, 0.f, 0.f, 0.f, 0.f, 0.f, 0.f, 0.f};
  return z;
}

// ---- CDNA5 async global->LDS copy (ASYNCcnt-tracked, no VGPR staging) ----
// ISA 10.7 / 15.18.3: GLOBAL_LOAD_ASYNC_TO_LDS_B128  vdst(lds addr), vaddr, off
// Flat->LDS address is a truncation to addr[31:0] (ISA 10.2 aperture table).
static __device__ __forceinline__ void async_copy_b128(void* lds, const void* g) {
  unsigned l = (unsigned)(unsigned long long)lds;
  asm volatile("global_load_async_to_lds_b128 %0, %1, off"
               :: "v"(l), "v"(g) : "memory");
}
static __device__ __forceinline__ void wait_async0() {
  asm volatile("s_wait_asynccnt 0x0" ::: "memory");
}

// ---------------- fp32 -> bf16 conversion ----------------
__global__ __launch_bounds__(256) void k_cvt_bf16(const float* __restrict__ src,
                                                  u16* __restrict__ dst, int n) {
  int i = (blockIdx.x * blockDim.x + threadIdx.x) * 4;
  if (i >= n) return;
  float4 f = *(const float4*)(src + i);
  uint2 r;
  r.x = (unsigned)bf16_of(f.x) | ((unsigned)bf16_of(f.y) << 16);
  r.y = (unsigned)bf16_of(f.z) | ((unsigned)bf16_of(f.w) << 16);
  *(uint2*)(dst + i) = r;
}

// ---------------- QKV GEMM + fused RoPE ----------------
// C[m,n] = sum_k x[m,k] * W[n,k];  block tile 64(M) x 128(N), K-step 32.
// Double-buffered LDS tiles filled by async global->LDS copies.
__global__ __launch_bounds__(256) void k_qkv_rope(
    const u16* __restrict__ xb,
    const u16* __restrict__ wq, const u16* __restrict__ wk, const u16* __restrict__ wv,
    u16* __restrict__ Qb, u16* __restrict__ Kb, u16* __restrict__ Vb)
{
  __shared__ __align__(16) u16 sA[2][64 * 32];    // x tile   [m][k]
  __shared__ __align__(16) u16 sB[2][128 * 32];   // W tile   [n][k]

  const int z = blockIdx.z;                       // 0=Q 1=K 2=V
  const u16* wb = (z == 0) ? wq : (z == 1) ? wk : wv;

  const int tid  = threadIdx.x;
  const int lane = tid & 31;
  const int wave = tid >> 5;
  const int hi   = (lane >> 4) & 1;
  const int l16  = lane & 15;
  const int wm   = wave & 3;
  const int wn   = wave >> 2;
  const int m0   = blockIdx.x * 64;
  const int n0   = blockIdx.y * 128;

  v8f acc[4];
#pragma unroll
  for (int j = 0; j < 4; ++j) acc[j] = vzero8();

  const int arow = tid >> 2, ac8  = (tid & 3) << 3;
  const int brow = tid >> 1, bc16 = (tid & 1) << 4;

  auto issue_tiles = [&](int k0, int buf) {
    async_copy_b128(&sA[buf][arow * 32 + ac8], &xb[(m0 + arow) * D_ + k0 + ac8]);
    const u16* bs = &wb[(n0 + brow) * D_ + k0 + bc16];
    async_copy_b128(&sB[buf][brow * 32 + bc16],     bs);
    async_copy_b128(&sB[buf][brow * 32 + bc16 + 8], bs + 8);
  };

  // prologue: fill buffer 0
  issue_tiles(0, 0);
  wait_async0();
  __syncthreads();

  int cur = 0;
  for (int k0 = 0; k0 < D_; k0 += 32) {
    if (k0 + 32 < D_) issue_tiles(k0 + 32, cur ^ 1);   // overlap copy with WMMA

    // A fragment: row = wm*16 + lane%16; lo lanes K 0..7 & 16..23, hi lanes +8.
    const u16* ap = &sA[cur][((wm << 4) + l16) * 32 + (hi ? 8 : 0)];
    v16bf af = frag_of(*(const uint4*)ap, *(const uint4*)(ap + 16));
    // Batch all B fragments, then issue WMMAs back-to-back.
    v16bf bfr[4];
#pragma unroll
    for (int j = 0; j < 4; ++j) {
      const u16* bp = &sB[cur][((wn << 6) + (j << 4) + l16) * 32 + (hi ? 8 : 0)];
      bfr[j] = frag_of(*(const uint4*)bp, *(const uint4*)(bp + 16));
    }
#pragma unroll
    for (int j = 0; j < 4; ++j)
      acc[j] = __builtin_amdgcn_wmma_f32_16x16x32_bf16(false, af, false, bfr[j],
                                                       (short)0, acc[j], false, false);

    wait_async0();        // next buffer landed
    __syncthreads();      // everyone done reading cur, next buffer published
    cur ^= 1;
  }

  // Epilogue: C layout row = v + 8*hi, col = lane%16. RoPE pairs (2i,2i+1) are
  // adjacent lanes -> partner via shfl_xor(1). Write head-major bf16.
#pragma unroll
  for (int j = 0; j < 4; ++j) {
#pragma unroll
    for (int v = 0; v < 8; ++v) {
      float val   = acc[j][v];
      float other = __shfl_xor(val, 1, 32);
      int col = n0 + (wn << 6) + (j << 4) + l16;
      int m   = m0 + (wm << 4) + v + (hi ? 8 : 0);
      int b = m >> 11, s = m & (S_ - 1);
      int h = col >> 6, ch = col & 63;
      int base = ((b * H_ + h) * S_ + s) * DH_;
      if (z == 2) {
        Vb[base + ch] = bf16_of(val);
      } else {
        float invf = __powf(10000.0f, -(float)(ch & ~1) * (1.0f / 64.0f));
        float ang = (float)s * invf;
        float sn = __sinf(ang), cs = __cosf(ang);
        float o; int chp;
        if ((ch & 1) == 0) { o = val * cs - other * sn; chp = ch >> 1; }        // x1*cos - x2*sin
        else               { o = other * sn + val * cs; chp = 32 + (ch >> 1); } // x1*sin + x2*cos
        u16* dst = (z == 0) ? Qb : Kb;
        dst[base + chp] = bf16_of(o);
      }
    }
  }
}

// ---------------- flash attention ----------------
// grid = (S/128, B*H); 8 waves, each owns 16 query rows; key blocks of 32.
__global__ __launch_bounds__(256) void k_attn(
    const u16* __restrict__ Qb, const u16* __restrict__ Kb, const u16* __restrict__ Vb,
    u16* __restrict__ ctx)
{
  __shared__ __align__(16) u16 sK [32 * 64];      // [key][ch]
  __shared__ __align__(16) u16 sVt[64 * 32];      // [ch][key] (transposed)
  __shared__ __align__(16) u16 sP [8 * 16 * 32];  // per-wave P transpose buffer

  const int tid = threadIdx.x, lane = tid & 31, wave = tid >> 5;
  const int hi = (lane >> 4) & 1, l16 = lane & 15;
  const int bh = blockIdx.y;
  const int b = bh >> 4, h = bh & 15;
  const int sq = blockIdx.x * 128 + wave * 16;

  const u16* Qh = Qb + (size_t)bh * S_ * DH_;
  const u16* Kh = Kb + (size_t)bh * S_ * DH_;
  const u16* Vh = Vb + (size_t)bh * S_ * DH_;

  // Q A-fragments for both 32-wide K-steps over Dh=64, held in registers.
  v16bf qf[2];
#pragma unroll
  for (int t = 0; t < 2; ++t) {
    const u16* qp = Qh + (sq + l16) * DH_ + (t << 5) + (hi ? 8 : 0);
    qf[t] = frag_of(*(const uint4*)qp, *(const uint4*)(qp + 16));
  }

  float mr[8], lr[8];
  v8f oacc[4];
#pragma unroll
  for (int v = 0; v < 8; ++v) { mr[v] = -3.0e38f; lr[v] = 0.f; }
#pragma unroll
  for (int nb = 0; nb < 4; ++nb) oacc[nb] = vzero8();

  const float isc = 0.125f;   // 1/sqrt(Dh)
  const int pb = wave * 512;
  const int kkey = tid >> 3, kc8 = (tid & 7) << 3;   // K-tile copy assignment
  const int vkey = tid & 31, vc0 = (tid >> 5) << 3;  // V-tile copy assignment

  for (int key0 = 0; key0 < S_; key0 += 32) {
    __syncthreads();                                 // all waves done with tiles
    // K tile via async global->LDS (32x64 row-major)
    async_copy_b128(&sK[kkey * 64 + kc8], &Kh[(key0 + kkey) * DH_ + kc8]);
    // V tile transposed: sVt[ch][key]
    {
      union { uint4 q; u16 s[8]; } u;
      u.q = *(const uint4*)&Vh[(key0 + vkey) * DH_ + vc0];
#pragma unroll
      for (int jj = 0; jj < 8; ++jj) sVt[(vc0 + jj) * 32 + vkey] = u.s[jj];
    }
    if (key0 + 32 < S_) {                            // prefetch next tiles
      __builtin_prefetch(&Kh[(key0 + 32 + kkey) * DH_ + kc8], 0, 1);
      __builtin_prefetch(&Vh[(key0 + 32 + vkey) * DH_ + vc0], 0, 1);
    }
    wait_async0();
    __syncthreads();

    // scores = Q(16x64) @ K^T(64x32): two 16-col tiles, two 32-deep K-steps.
    v16bf kf[4];                                     // [t*2 + j]
#pragma unroll
    for (int t = 0; t < 2; ++t) {
#pragma unroll
      for (int j = 0; j < 2; ++j) {
        const u16* bp = &sK[((j << 4) + l16) * 64 + (t << 5) + (hi ? 8 : 0)];
        kf[t * 2 + j] = frag_of(*(const uint4*)bp, *(const uint4*)(bp + 16));
      }
    }
    v8f s0 = vzero8(), s1 = vzero8();
    s0 = __builtin_amdgcn_wmma_f32_16x16x32_bf16(false, qf[0], false, kf[0], (short)0, s0, false, false);
    s1 = __builtin_amdgcn_wmma_f32_16x16x32_bf16(false, qf[0], false, kf[1], (short)0, s1, false, false);
    s0 = __builtin_amdgcn_wmma_f32_16x16x32_bf16(false, qf[1], false, kf[2], (short)0, s0, false, false);
    s1 = __builtin_amdgcn_wmma_f32_16x16x32_bf16(false, qf[1], false, kf[3], (short)0, s1, false, false);

    // online softmax: rows live in (v, lane-half); cols across 16 lanes.
#pragma unroll
    for (int v = 0; v < 8; ++v) {
      float x0 = s0[v] * isc, x1 = s1[v] * isc;
      float mx = fmaxf(x0, x1);
#pragma unroll
      for (int off = 1; off < 16; off <<= 1) mx = fmaxf(mx, __shfl_xor(mx, off, 32));
      float mnew  = fmaxf(mr[v], mx);
      float alpha = __expf(mr[v] - mnew);
      float p0 = __expf(x0 - mnew), p1 = __expf(x1 - mnew);
      float rs = p0 + p1;
#pragma unroll
      for (int off = 1; off < 16; off <<= 1) rs += __shfl_xor(rs, off, 32);
      lr[v] = lr[v] * alpha + rs;
      mr[v] = mnew;
#pragma unroll
      for (int nb = 0; nb < 4; ++nb) oacc[nb][v] *= alpha;
      int r = v + (hi ? 8 : 0);
      sP[pb + r * 32 + l16]      = bf16_of(p0);
      sP[pb + r * 32 + 16 + l16] = bf16_of(p1);
    }

    // O += P(16x32) @ V(32x64). P re-read in A-fragment layout (per-wave DS ops
    // are in-order, so no barrier needed for the private sP buffer).
    const u16* pp = &sP[pb + l16 * 32 + (hi ? 8 : 0)];
    v16bf pf = frag_of(*(const uint4*)pp, *(const uint4*)(pp + 16));
    v16bf vf[4];
#pragma unroll
    for (int nb = 0; nb < 4; ++nb) {
      const u16* vp = &sVt[((nb << 4) + l16) * 32 + (hi ? 8 : 0)];
      vf[nb] = frag_of(*(const uint4*)vp, *(const uint4*)(vp + 16));
    }
#pragma unroll
    for (int nb = 0; nb < 4; ++nb)
      oacc[nb] = __builtin_amdgcn_wmma_f32_16x16x32_bf16(false, pf, false, vf[nb],
                                                         (short)0, oacc[nb], false, false);
  }

  // ctx[b, s, h*64 + ch] = O / l
#pragma unroll
  for (int nb = 0; nb < 4; ++nb) {
#pragma unroll
    for (int v = 0; v < 8; ++v) {
      float val = oacc[nb][v] / lr[v];
      int srow  = sq + v + (hi ? 8 : 0);
      int chg   = (h << 6) + (nb << 4) + l16;
      ctx[(size_t)(b * S_ + srow) * D_ + chg] = bf16_of(val);
    }
  }
}

// ---------------- output projection: out = ctx @ Wo^T (fp32 out) ----------------
__global__ __launch_bounds__(256) void k_oproj(
    const u16* __restrict__ cb, const u16* __restrict__ wo, float* __restrict__ out)
{
  __shared__ __align__(16) u16 sA[2][64 * 32];
  __shared__ __align__(16) u16 sB[2][128 * 32];

  const int tid  = threadIdx.x;
  const int lane = tid & 31;
  const int wave = tid >> 5;
  const int hi   = (lane >> 4) & 1;
  const int l16  = lane & 15;
  const int wm   = wave & 3;
  const int wn   = wave >> 2;
  const int m0   = blockIdx.x * 64;
  const int n0   = blockIdx.y * 128;

  v8f acc[4];
#pragma unroll
  for (int j = 0; j < 4; ++j) acc[j] = vzero8();

  const int arow = tid >> 2, ac8  = (tid & 3) << 3;
  const int brow = tid >> 1, bc16 = (tid & 1) << 4;

  auto issue_tiles = [&](int k0, int buf) {
    async_copy_b128(&sA[buf][arow * 32 + ac8], &cb[(m0 + arow) * D_ + k0 + ac8]);
    const u16* bs = &wo[(n0 + brow) * D_ + k0 + bc16];
    async_copy_b128(&sB[buf][brow * 32 + bc16],     bs);
    async_copy_b128(&sB[buf][brow * 32 + bc16 + 8], bs + 8);
  };

  issue_tiles(0, 0);
  wait_async0();
  __syncthreads();

  int cur = 0;
  for (int k0 = 0; k0 < D_; k0 += 32) {
    if (k0 + 32 < D_) issue_tiles(k0 + 32, cur ^ 1);

    const u16* ap = &sA[cur][((wm << 4) + l16) * 32 + (hi ? 8 : 0)];
    v16bf af = frag_of(*(const uint4*)ap, *(const uint4*)(ap + 16));
    v16bf bfr[4];
#pragma unroll
    for (int j = 0; j < 4; ++j) {
      const u16* bp = &sB[cur][((wn << 6) + (j << 4) + l16) * 32 + (hi ? 8 : 0)];
      bfr[j] = frag_of(*(const uint4*)bp, *(const uint4*)(bp + 16));
    }
#pragma unroll
    for (int j = 0; j < 4; ++j)
      acc[j] = __builtin_amdgcn_wmma_f32_16x16x32_bf16(false, af, false, bfr[j],
                                                       (short)0, acc[j], false, false);

    wait_async0();
    __syncthreads();
    cur ^= 1;
  }

#pragma unroll
  for (int j = 0; j < 4; ++j) {
#pragma unroll
    for (int v = 0; v < 8; ++v) {
      int col = n0 + (wn << 6) + (j << 4) + l16;
      int m   = m0 + (wm << 4) + v + (hi ? 8 : 0);
      out[(size_t)m * D_ + col] = acc[j][v];
    }
  }
}

extern "C" void kernel_launch(void* const* d_in, const int* in_sizes, int n_in,
                              void* d_out, int out_size, void* d_ws, size_t ws_size,
                              hipStream_t stream) {
  (void)in_sizes; (void)n_in; (void)out_size; (void)ws_size;
  const float* x  = (const float*)d_in[0];
  const float* Wq = (const float*)d_in[1];
  const float* Wk = (const float*)d_in[2];
  const float* Wv = (const float*)d_in[3];
  const float* Wo = (const float*)d_in[4];

  char* ws = (char*)d_ws;
  const size_t MB = 1024 * 1024;
  u16* XB  = (u16*)(ws + 0 * MB);
  u16* WQB = (u16*)(ws + 8 * MB);
  u16* WKB = (u16*)(ws + 10 * MB);
  u16* WVB = (u16*)(ws + 12 * MB);
  u16* WOB = (u16*)(ws + 14 * MB);
  u16* Qb  = (u16*)(ws + 16 * MB);
  u16* Kb  = (u16*)(ws + 24 * MB);
  u16* Vb  = (u16*)(ws + 32 * MB);
  u16* CT  = (u16*)(ws + 40 * MB);

  k_cvt_bf16<<<4096, 256, 0, stream>>>(x,  XB,  B_ * S_ * D_);
  k_cvt_bf16<<<1024, 256, 0, stream>>>(Wq, WQB, D_ * D_);
  k_cvt_bf16<<<1024, 256, 0, stream>>>(Wk, WKB, D_ * D_);
  k_cvt_bf16<<<1024, 256, 0, stream>>>(Wv, WVB, D_ * D_);
  k_cvt_bf16<<<1024, 256, 0, stream>>>(Wo, WOB, D_ * D_);

  k_qkv_rope<<<dim3(64, 8, 3), 256, 0, stream>>>(XB, WQB, WKB, WVB, Qb, Kb, Vb);
  k_attn   <<<dim3(16, 32),    256, 0, stream>>>(Qb, Kb, Vb, CT);
  k_oproj  <<<dim3(64, 8),     256, 0, stream>>>(CT, WOB, (float*)d_out);
}
